// EGNN_80376017977785
// MI455X (gfx1250) — matrix operand order repaired and checked by reading, not compiled, not run on hardware
//
#include <hip/hip_runtime.h>
#include <math.h>

#define NN   25000
#define EE   400000
#define BB   64
#define HIDD 128
#define LAY  4

typedef __attribute__((ext_vector_type(16))) __bf16 bfx16;
typedef __attribute__((ext_vector_type(8)))  __bf16 bfx8;
typedef __attribute__((ext_vector_type(8)))  float  fx8;

__device__ __forceinline__ float siluf(float v) { return v / (1.f + __expf(-v)); }

// ---------------------------------------------------------------------------
// Wave-level GEMM (single M tile): D(16 x NT*16) += A(16 x kd) * B(kd x NT*16)
// A: LDS bf16 row-major [16][astride]; Bt: LDS bf16 col-major (Bt[n][k]).
// ---------------------------------------------------------------------------
template <int NT>
__device__ __forceinline__ void wave_gemm(const __bf16* A, int astride,
                                          const __bf16* Bt, int bstride,
                                          int kd, fx8* acc, int lane)
{
    const int ln   = lane & 15;
    const int half = lane >> 4;
    for (int kk = 0; kk < kd; kk += 32) {
        const __bf16* rp = A + ln * astride + kk + half * 16;
        bfx8 a0 = *(const bfx8*)(rp);
        bfx8 a1 = *(const bfx8*)(rp + 8);
        bfx16 af = __builtin_shufflevector(a0, a1,
                       0,1,2,3,4,5,6,7,8,9,10,11,12,13,14,15);
#pragma unroll
        for (int t = 0; t < NT; ++t) {
            const __bf16* bp = Bt + (t * 16 + ln) * bstride + kk + half * 16;
            bfx16 bf = *(const bfx16*)(bp);
            acc[t] = __builtin_amdgcn_wmma_f32_16x16x32_bf16(
                         false, af, false, bf, (short)0, acc[t], false, false);
        }
    }
}

// ---------------------------------------------------------------------------
// Wave-level GEMM, 2 M-tiles (32 rows): each B fragment feeds 2 independent
// WMMAs -> halves LDS traffic per MMA and gives the scheduler ILP to cover
// ds_load latency.
// ---------------------------------------------------------------------------
template <int NT>
__device__ __forceinline__ void wave_gemm2(const __bf16* A, int astride,
                                           const __bf16* Bt, int bstride,
                                           int kd, fx8 (*acc)[NT], int lane)
{
    const int ln   = lane & 15;
    const int half = lane >> 4;
    for (int kk = 0; kk < kd; kk += 32) {
        const __bf16* rp0 = A + ln * astride + kk + half * 16;
        const __bf16* rp1 = rp0 + 16 * astride;
        bfx8 a0 = *(const bfx8*)(rp0);
        bfx8 a1 = *(const bfx8*)(rp0 + 8);
        bfx8 a2 = *(const bfx8*)(rp1);
        bfx8 a3 = *(const bfx8*)(rp1 + 8);
        bfx16 af0 = __builtin_shufflevector(a0, a1,
                        0,1,2,3,4,5,6,7,8,9,10,11,12,13,14,15);
        bfx16 af1 = __builtin_shufflevector(a2, a3,
                        0,1,2,3,4,5,6,7,8,9,10,11,12,13,14,15);
#pragma unroll
        for (int t = 0; t < NT; ++t) {
            const __bf16* bp = Bt + (t * 16 + ln) * bstride + kk + half * 16;
            bfx16 bf = *(const bfx16*)(bp);
            acc[0][t] = __builtin_amdgcn_wmma_f32_16x16x32_bf16(
                            false, af0, false, bf, (short)0, acc[0][t], false, false);
            acc[1][t] = __builtin_amdgcn_wmma_f32_16x16x32_bf16(
                            false, af1, false, bf, (short)0, acc[1][t], false, false);
        }
    }
}

template <int NT>
__device__ __forceinline__ void zero_acc(fx8* acc)
{
#pragma unroll
    for (int t = 0; t < NT; ++t)
#pragma unroll
        for (int j = 0; j < 8; ++j) acc[t][j] = 0.f;
}

// Store D tiles (+bias, optional SiLU) back to LDS row-major bf16.
template <int NT>
__device__ __forceinline__ void store_act_bf16(const fx8* acc, __bf16* out,
                                               int ostride, int coloff,
                                               const float* bias, bool act,
                                               int lane)
{
    const int ln   = lane & 15;
    const int half = lane >> 4;
#pragma unroll
    for (int t = 0; t < NT; ++t)
#pragma unroll
        for (int r = 0; r < 8; ++r) {
            int rowi = r + 8 * half;
            int c    = t * 16 + ln;
            float v  = acc[t][r] + bias[c];
            if (act) v = siluf(v);
            out[rowi * ostride + coloff + c] = (__bf16)v;
        }
}

// ---------------------------------------------------------------------------
// Time embedding (tiny, VALU)
// ---------------------------------------------------------------------------
__global__ void time_kernel(const float* __restrict__ t,
                            const float* __restrict__ w1, const float* __restrict__ b1,
                            const float* __restrict__ w2, const float* __restrict__ b2,
                            float* __restrict__ te2)
{
    __shared__ float se[64];
    __shared__ float hid[128];
    const int b = blockIdx.x, tid = threadIdx.x;
    if (tid < 32) {
        float f = __expf((float)tid * (-logf(10000.f) / 31.f));
        float a = t[b] * f;
        se[tid]      = __sinf(a);
        se[tid + 32] = __cosf(a);
    }
    __syncthreads();
    float s = b1[tid];
    for (int k = 0; k < 64; ++k) s += se[k] * w1[k * 128 + tid];
    hid[tid] = siluf(s);
    __syncthreads();
    float o = b2[tid];
    for (int k = 0; k < 128; ++k) o += hid[k] * w2[k * 128 + tid];
    te2[b * 128 + tid] = o;
}

// h0 = h@node_w + node_b + te2[batch]
__global__ void embed_kernel(const float* __restrict__ h16,
                             const int* __restrict__ batch,
                             const float* __restrict__ nodew,
                             const float* __restrict__ nodeb,
                             const float* __restrict__ te2,
                             float* __restrict__ hout)
{
    int i = blockIdx.x * blockDim.x + threadIdx.x;
    if (i >= NN * HIDD) return;
    int n = i >> 7, c = i & 127;
    float s = nodeb[c];
    const float* hp = h16 + (size_t)n * 16;
#pragma unroll
    for (int k = 0; k < 16; ++k) s += hp[k] * nodew[k * 128 + c];
    hout[i] = s + te2[batch[n] * 128 + c];
}

// ---------------------------------------------------------------------------
// Edge kernel: 8 waves x 32 edges. Three WMMA GEMM chains (2 M-tiles each),
// attention / coordinate-weight dots, atomic segment sums.
// ---------------------------------------------------------------------------
__global__ __launch_bounds__(256, 1) void edge_kernel(
    const float* __restrict__ h, const float* __restrict__ xin,
    const int* __restrict__ erow, const int* __restrict__ ecol,
    const float* __restrict__ eattr,
    const float* __restrict__ w1, const float* __restrict__ b1,
    const float* __restrict__ w2, const float* __restrict__ b2,
    const float* __restrict__ aw, const float* __restrict__ ab,
    const float* __restrict__ c1, const float* __restrict__ cb,
    const float* __restrict__ c2,
    float* __restrict__ msum, float* __restrict__ xout)
{
    __shared__ alignas(32) __bf16 sW1[128 * 288];   // edge_w1^T (K padded 273->288)
    __shared__ alignas(32) __bf16 sW2[128 * 128];   // edge_w2^T
    __shared__ alignas(32) __bf16 sC1[128 * 128];   // cw1^T
    __shared__ alignas(32) __bf16 rowbuf[8][32 * 288];
    __shared__ float sB1[128], sB2[128], sCB[128], sAW[128], sC2[128];
    __shared__ float sAB;
    __shared__ float cdb[8][32][4];
    __shared__ float radf[8][32];
    __shared__ float attv[8][32];
    __shared__ float cwv[8][32];

    const int tid = threadIdx.x;
    for (int idx = tid; idx < 128 * 288; idx += 256) {
        int n = idx / 288, k = idx - n * 288;
        sW1[idx] = (__bf16)((k < 273) ? w1[k * 128 + n] : 0.f);
    }
    for (int idx = tid; idx < 128 * 128; idx += 256) {
        int n = idx >> 7, k = idx & 127;
        sW2[idx] = (__bf16)w2[k * 128 + n];
        sC1[idx] = (__bf16)c1[k * 128 + n];
    }
    if (tid < 128) {
        sB1[tid] = b1[tid]; sB2[tid] = b2[tid]; sCB[tid] = cb[tid];
        sAW[tid] = aw[tid]; sC2[tid] = c2[tid];
    }
    if (tid == 0) sAB = ab[0];
    __syncthreads();

    const int wave = tid >> 5, lane = tid & 31;
    const long long ebase = (long long)blockIdx.x * 256 + wave * 32;
    __bf16* rb = rowbuf[wave];
    const int c0 = lane * 4;

    // ---- stage edge_in = [h[row] | h[col] | radial | edge_attr | pad] as bf16
    for (int e = 0; e < 32; ++e) {
        long long eg = ebase + e;
        bool ok = eg < (long long)EE;
        int r  = ok ? erow[eg] : 0;
        int cl = ok ? ecol[eg] : 0;
        float4 z4 = make_float4(0.f, 0.f, 0.f, 0.f);
        float4 hr = ok ? *reinterpret_cast<const float4*>(h + (size_t)r  * 128 + c0) : z4;
        float4 hc = ok ? *reinterpret_cast<const float4*>(h + (size_t)cl * 128 + c0) : z4;
        __bf16* rp = rb + e * 288;
        rp[c0 + 0] = (__bf16)hr.x; rp[c0 + 1] = (__bf16)hr.y;
        rp[c0 + 2] = (__bf16)hr.z; rp[c0 + 3] = (__bf16)hr.w;
        rp[128 + c0 + 0] = (__bf16)hc.x; rp[128 + c0 + 1] = (__bf16)hc.y;
        rp[128 + c0 + 2] = (__bf16)hc.z; rp[128 + c0 + 3] = (__bf16)hc.w;
        if (lane < 3)
            cdb[wave][e][lane] = ok ? (xin[(size_t)r * 3 + lane] - xin[(size_t)cl * 3 + lane]) : 0.f;
        if (lane < 16)
            rp[257 + lane] = (__bf16)(ok ? eattr[eg * 16 + lane] : 0.f);
        if (lane >= 16 && lane < 31)
            rp[273 + (lane - 16)] = (__bf16)0.f;
    }
    __syncthreads();
    {   // radial: one edge per lane
        float d0 = cdb[wave][lane][0], d1 = cdb[wave][lane][1], d2 = cdb[wave][lane][2];
        float rad = d0 * d0 + d1 * d1 + d2 * d2;
        radf[wave][lane] = rad;
        rb[lane * 288 + 256] = (__bf16)rad;
    }
    __syncthreads();

    fx8 acc[2][8];
    // ---- GEMM1: edge_in(32x288) @ W1 -> m1 (silu), into cols [0,128)
    zero_acc<8>(acc[0]); zero_acc<8>(acc[1]);
    wave_gemm2<8>(rb, 288, sW1, 288, 288, acc, lane);
    __syncthreads();
    store_act_bf16<8>(acc[0], rb,            288, 0, sB1, true, lane);
    store_act_bf16<8>(acc[1], rb + 16 * 288, 288, 0, sB1, true, lane);
    __syncthreads();
    // ---- GEMM2: m1 @ W2 -> m (silu), into cols [128,256)
    zero_acc<8>(acc[0]); zero_acc<8>(acc[1]);
    wave_gemm2<8>(rb, 288, sW2, 128, 128, acc, lane);
    __syncthreads();
    store_act_bf16<8>(acc[0], rb,            288, 128, sB2, true, lane);
    store_act_bf16<8>(acc[1], rb + 16 * 288, 288, 128, sB2, true, lane);
    __syncthreads();
    // ---- attention: sigmoid(m @ att_w + att_b); one edge per lane
    {
        const __bf16* mp = rb + lane * 288 + 128;
        float s = sAB;
        for (int c = 0; c < 128; ++c) s += (float)mp[c] * sAW[c];
        attv[wave][lane] = 1.f / (1.f + __expf(-s));
    }
    __syncthreads();
    for (int i = lane; i < 32 * 128; i += 32) {
        int e = i >> 7, c = i & 127;
        rb[e * 288 + 128 + c] = (__bf16)((float)rb[e * 288 + 128 + c] * attv[wave][e]);
    }
    __syncthreads();
    // ---- GEMM3: m @ C1 -> ch (silu), into cols [0,128)
    zero_acc<8>(acc[0]); zero_acc<8>(acc[1]);
    wave_gemm2<8>(rb + 128, 288, sC1, 128, 128, acc, lane);
    __syncthreads();
    store_act_bf16<8>(acc[0], rb,            288, 0, sCB, true, lane);
    store_act_bf16<8>(acc[1], rb + 16 * 288, 288, 0, sCB, true, lane);
    __syncthreads();
    // ---- cw = ch @ c2 (128 -> 1); one edge per lane
    {
        const __bf16* cp = rb + lane * 288;
        float s = 0.f;
        for (int c = 0; c < 128; ++c) s += (float)cp[c] * sC2[c];
        cwv[wave][lane] = s;
    }
    __syncthreads();
    // ---- coordinate update: x += segsum(coord_diff/sqrt(rad+eps) * cw)
    {
        long long eg = ebase + lane;
        if (eg < (long long)EE) {
            int r = erow[eg];
            float inv = rsqrtf(radf[wave][lane] + 1e-8f);
            float s = cwv[wave][lane] * inv;
#pragma unroll
            for (int d = 0; d < 3; ++d)
                atomicAdd(&xout[(size_t)r * 3 + d], cdb[wave][lane][d] * s);
        }
    }
    // ---- message aggregation: m_i = segsum(m)
    for (int e = 0; e < 32; ++e) {
        long long eg = ebase + e;
        if (eg >= (long long)EE) break;
        int r = erow[eg];
        const __bf16* mp = rb + e * 288 + 128 + c0;
        float* dst = msum + (size_t)r * 128 + c0;
        atomicAdd(dst + 0, (float)mp[0]);
        atomicAdd(dst + 1, (float)mp[1]);
        atomicAdd(dst + 2, (float)mp[2]);
        atomicAdd(dst + 3, (float)mp[3]);
    }
}

// ---------------------------------------------------------------------------
// Node kernel: h = LN(h + MLP([h | m_i]))
// ---------------------------------------------------------------------------
__global__ __launch_bounds__(256, 1) void node_kernel(
    const float* __restrict__ hin, const float* __restrict__ msum,
    const float* __restrict__ w1, const float* __restrict__ b1,
    const float* __restrict__ w2, const float* __restrict__ b2,
    const float* __restrict__ g, const float* __restrict__ be,
    float* __restrict__ hout)
{
    __shared__ alignas(32) __bf16 sW1[128 * 256];
    __shared__ alignas(32) __bf16 sW2[128 * 128];
    __shared__ alignas(32) __bf16 rbuf[8][16 * 256];
    __shared__ float rf[8][16][128];
    __shared__ float sB1[128], sB2[128], sG[128], sBe[128];
    __shared__ float mu[8][16], iv[8][16];

    const int tid = threadIdx.x;
    for (int idx = tid; idx < 128 * 256; idx += 256) {
        int n = idx >> 8, k = idx & 255;
        sW1[idx] = (__bf16)w1[k * 128 + n];
    }
    for (int idx = tid; idx < 128 * 128; idx += 256) {
        int n = idx >> 7, k = idx & 127;
        sW2[idx] = (__bf16)w2[k * 128 + n];
    }
    if (tid < 128) { sB1[tid] = b1[tid]; sB2[tid] = b2[tid]; sG[tid] = g[tid]; sBe[tid] = be[tid]; }
    __syncthreads();

    const int wave = tid >> 5, lane = tid & 31;
    const int nbase = blockIdx.x * 128 + wave * 16;
    __bf16* rb = rbuf[wave];
    const int c0 = lane * 4;

    for (int e = 0; e < 16; ++e) {
        int node = nbase + e;
        bool ok = node < NN;
        float4 z4 = make_float4(0.f, 0.f, 0.f, 0.f);
        float4 hv = ok ? *reinterpret_cast<const float4*>(hin  + (size_t)node * 128 + c0) : z4;
        float4 mv = ok ? *reinterpret_cast<const float4*>(msum + (size_t)node * 128 + c0) : z4;
        __bf16* rp = rb + e * 256;
        rp[c0 + 0] = (__bf16)hv.x; rp[c0 + 1] = (__bf16)hv.y;
        rp[c0 + 2] = (__bf16)hv.z; rp[c0 + 3] = (__bf16)hv.w;
        rp[128 + c0 + 0] = (__bf16)mv.x; rp[128 + c0 + 1] = (__bf16)mv.y;
        rp[128 + c0 + 2] = (__bf16)mv.z; rp[128 + c0 + 3] = (__bf16)mv.w;
    }
    __syncthreads();

    fx8 acc[8];
    zero_acc<8>(acc);
    wave_gemm<8>(rb, 256, sW1, 256, 256, acc, lane);
    __syncthreads();
    store_act_bf16<8>(acc, rb, 256, 0, sB1, true, lane);
    __syncthreads();
    zero_acc<8>(acc);
    wave_gemm<8>(rb, 256, sW2, 128, 128, acc, lane);
    __syncthreads();
    {
        const int ln = lane & 15, half = lane >> 4;
#pragma unroll
        for (int t = 0; t < 8; ++t)
#pragma unroll
            for (int r = 0; r < 8; ++r) {
                int rowi = r + 8 * half, c = t * 16 + ln;
                int node = nbase + rowi;
                float res = (node < NN) ? hin[(size_t)node * 128 + c] : 0.f;
                rf[wave][rowi][c] = acc[t][r] + sB2[c] + res;
            }
    }
    __syncthreads();
    if (lane < 16) {
        float s = 0.f, s2 = 0.f;
        for (int c = 0; c < 128; ++c) { float v = rf[wave][lane][c]; s += v; s2 += v * v; }
        float m_ = s * (1.f / 128.f);
        float var = s2 * (1.f / 128.f) - m_ * m_;
        mu[wave][lane] = m_;
        iv[wave][lane] = rsqrtf(var + 1e-5f);
    }
    __syncthreads();
    for (int i = lane; i < 16 * 128; i += 32) {
        int e = i >> 7, c = i & 127;
        int node = nbase + e;
        if (node < NN)
            hout[(size_t)node * 128 + c] =
                (rf[wave][e][c] - mu[wave][e]) * iv[wave][e] * sG[c] + sBe[c];
    }
}

// ---------------------------------------------------------------------------
// Output heads: h_out = silu(h@W1+b1)@W2+b2 ; x_out = h@head_w + head_b
// ---------------------------------------------------------------------------
__global__ __launch_bounds__(256, 1) void head_kernel(
    const float* __restrict__ hin,
    const float* __restrict__ w1, const float* __restrict__ b1,
    const float* __restrict__ w2, const float* __restrict__ b2,
    const float* __restrict__ hw, const float* __restrict__ hb,
    float* __restrict__ out_h, float* __restrict__ out_x)
{
    __shared__ alignas(32) __bf16 sW1[128 * 128];
    __shared__ alignas(32) __bf16 sW2[16 * 128];
    __shared__ alignas(32) __bf16 rbuf[8][16 * 128];
    __shared__ float sB1[128], sHW[384], sB2[16], sHB[3];

    const int tid = threadIdx.x;
    for (int idx = tid; idx < 128 * 128; idx += 256) {
        int n = idx >> 7, k = idx & 127;
        sW1[idx] = (__bf16)w1[k * 128 + n];
    }
    for (int idx = tid; idx < 16 * 128; idx += 256) {
        int n = idx >> 7, k = idx & 127;
        sW2[idx] = (__bf16)w2[k * 16 + n];
    }
    for (int idx = tid; idx < 384; idx += 256) {
        int d = idx / 128, c = idx - d * 128;
        sHW[idx] = hw[c * 3 + d];
    }
    if (tid < 128) sB1[tid] = b1[tid];
    if (tid < 16)  sB2[tid] = b2[tid];
    if (tid < 3)   sHB[tid] = hb[tid];
    __syncthreads();

    const int wave = tid >> 5, lane = tid & 31;
    const int nbase = blockIdx.x * 128 + wave * 16;
    __bf16* rb = rbuf[wave];
    const int c0 = lane * 4;

    for (int e = 0; e < 16; ++e) {
        int node = nbase + e;
        bool ok = node < NN;
        float4 z4 = make_float4(0.f, 0.f, 0.f, 0.f);
        float4 hv = ok ? *reinterpret_cast<const float4*>(hin + (size_t)node * 128 + c0) : z4;
        __bf16* rp = rb + e * 128;
        rp[c0 + 0] = (__bf16)hv.x; rp[c0 + 1] = (__bf16)hv.y;
        rp[c0 + 2] = (__bf16)hv.z; rp[c0 + 3] = (__bf16)hv.w;
    }
    __syncthreads();
    // coordinate head (3-wide): per-lane dots
    if (lane < 16) {
        int node = nbase + lane;
        if (node < NN) {
            const __bf16* hp = rb + lane * 128;
#pragma unroll
            for (int d = 0; d < 3; ++d) {
                float s = sHB[d];
                for (int c = 0; c < 128; ++c) s += (float)hp[c] * sHW[d * 128 + c];
                out_x[(size_t)node * 3 + d] = s;
            }
        }
    }
    __syncthreads();
    fx8 acc[8];
    zero_acc<8>(acc);
    wave_gemm<8>(rb, 128, sW1, 128, 128, acc, lane);
    __syncthreads();
    store_act_bf16<8>(acc, rb, 128, 0, sB1, true, lane);
    __syncthreads();
    fx8 acc1[1];
    zero_acc<1>(acc1);
    wave_gemm<1>(rb, 128, sW2, 128, 128, acc1, lane);
    {
        const int ln = lane & 15, half = lane >> 4;
#pragma unroll
        for (int r = 0; r < 8; ++r) {
            int rowi = r + 8 * half;
            int node = nbase + rowi;
            if (node < NN) out_h[(size_t)node * 16 + ln] = acc1[0][r] + sB2[ln];
        }
    }
}

// ---------------------------------------------------------------------------
extern "C" void kernel_launch(void* const* d_in, const int* in_sizes, int n_in,
                              void* d_out, int out_size, void* d_ws, size_t ws_size,
                              hipStream_t stream)
{
    (void)in_sizes; (void)n_in; (void)out_size; (void)ws_size;

    const float* h16   = (const float*)d_in[0];
    const float* x0    = (const float*)d_in[1];
    const int*   eidx  = (const int*)d_in[2];
    const float* tt    = (const float*)d_in[3];
    const float* eat   = (const float*)d_in[4];
    const int*   batch = (const int*)d_in[5];
    const float* nodew = (const float*)d_in[6];
    const float* nodeb = (const float*)d_in[7];
    const float* tw1   = (const float*)d_in[8];
    const float* tb1   = (const float*)d_in[9];
    const float* tw2   = (const float*)d_in[10];
    const float* tb2   = (const float*)d_in[11];
    const float* ew1   = (const float*)d_in[12];
    const float* eb1   = (const float*)d_in[13];
    const float* ew2   = (const float*)d_in[14];
    const float* eb2   = (const float*)d_in[15];
    const float* aw    = (const float*)d_in[16];
    const float* ab    = (const float*)d_in[17];
    const float* c1w   = (const float*)d_in[18];
    const float* c1b   = (const float*)d_in[19];
    const float* c2w   = (const float*)d_in[20];
    const float* nw1   = (const float*)d_in[21];
    const float* nb1   = (const float*)d_in[22];
    const float* nw2   = (const float*)d_in[23];
    const float* nb2   = (const float*)d_in[24];
    const float* lng   = (const float*)d_in[25];
    const float* lnb   = (const float*)d_in[26];
    const float* ow1   = (const float*)d_in[27];
    const float* ob1   = (const float*)d_in[28];
    const float* ow2   = (const float*)d_in[29];
    const float* ob2   = (const float*)d_in[30];
    const float* hw    = (const float*)d_in[31];
    const float* hb    = (const float*)d_in[32];

    float* ws  = (float*)d_ws;
    float* te2 = ws;
    float* hA  = te2 + BB * HIDD;
    float* hB  = hA + (size_t)NN * HIDD;
    float* ms  = hB + (size_t)NN * HIDD;
    float* xA  = ms + (size_t)NN * HIDD;
    float* xB  = xA + (size_t)NN * 3;

    const int* erow = eidx;
    const int* ecol = eidx + EE;

    hipMemcpyAsync(xA, x0, (size_t)NN * 3 * sizeof(float),
                   hipMemcpyDeviceToDevice, stream);
    time_kernel<<<BB, 128, 0, stream>>>(tt, tw1, tb1, tw2, tb2, te2);
    embed_kernel<<<(NN * HIDD + 255) / 256, 256, 0, stream>>>(
        h16, batch, nodew, nodeb, te2, hA);

    float* hin = hA;  float* hout = hB;
    float* xin = xA;  float* xout = xB;
    for (int i = 0; i < LAY; ++i) {
        hipMemsetAsync(ms, 0, (size_t)NN * HIDD * sizeof(float), stream);
        hipMemcpyAsync(xout, xin, (size_t)NN * 3 * sizeof(float),
                       hipMemcpyDeviceToDevice, stream);
        edge_kernel<<<(EE + 255) / 256, 256, 0, stream>>>(
            hin, xin, erow, ecol, eat,
            ew1 + (size_t)i * 273 * 128, eb1 + i * 128,
            ew2 + (size_t)i * 128 * 128, eb2 + i * 128,
            aw + i * 128, ab + i,
            c1w + (size_t)i * 128 * 128, c1b + i * 128, c2w + i * 128,
            ms, xout);
        node_kernel<<<(NN + 127) / 128, 256, 0, stream>>>(
            hin, ms,
            nw1 + (size_t)i * 256 * 128, nb1 + i * 128,
            nw2 + (size_t)i * 128 * 128, nb2 + i * 128,
            lng + i * 128, lnb + i * 128, hout);
        float* th = hin; hin = hout; hout = th;
        float* tx = xin; xin = xout; xout = tx;
    }

    head_kernel<<<(NN + 127) / 128, 256, 0, stream>>>(
        hin, ow1, ob1, ow2, ob2, hw, hb,
        (float*)d_out, (float*)d_out + (size_t)NN * 16);
}
